// ASMLoc_Base_79413945303623
// MI455X (gfx1250) — matrix operand order repaired and checked by reading, not compiled, not run on hardware
//
#include <hip/hip_runtime.h>
#include <cstdint>
#include <cstddef>

#define Bb 32
#define Tt 1024
#define Dd 2048
#define NC 21

// output offsets (flat concat in reference return order)
#define OFF_FG   0
#define OFF_BG   672
#define OFF_ATT  1344
#define OFF_CAS  66880
#define OFF_FGC  755008
#define OFF_BGC  1443136

typedef __attribute__((ext_vector_type(16))) __bf16 v16bf;
typedef __attribute__((ext_vector_type(8)))  float  v8f;
typedef int v4i __attribute__((vector_size(16)));          // matches builtin param
typedef __attribute__((address_space(1))) v4i gv4i;        // global int4
typedef __attribute__((address_space(3))) v4i lv4i;        // LDS int4

union Frag32 { uint4 u[2]; v16bf v; };
union BF8Pack { __bf16 h[8]; uint4 u; };

// ---- gfx1250 async-to-LDS path (guarded; falls back to plain ld/st) -------
#if defined(__has_builtin)
#  if __has_builtin(__builtin_amdgcn_global_load_async_to_lds_b128)
#    define HAVE_ASYNC_LDS 1
#  endif
#  if __has_builtin(__builtin_amdgcn_s_wait_asynccnt)
#    define HAVE_WAIT_ASYNC 1
#  endif
#endif

__device__ __forceinline__ void wait_async_zero() {
#if defined(HAVE_ASYNC_LDS)
#  if defined(HAVE_WAIT_ASYNC)
  __builtin_amdgcn_s_wait_asynccnt(0);
#  else
  asm volatile("s_wait_asynccnt 0x0" ::: "memory");
#  endif
#endif
}

// ---------------------------------------------------------------------------
// 1) cast input_feature f32 -> bf16 (8 elems per thread)
// ---------------------------------------------------------------------------
__global__ void k_cast_x(const float* __restrict__ x, unsigned short* __restrict__ xb) {
  size_t i = (size_t)blockIdx.x * blockDim.x + threadIdx.x;  // 8M threads
  const float4* xin = reinterpret_cast<const float4*>(x);
  float4 a = xin[i * 2 + 0];
  float4 b = xin[i * 2 + 1];
  BF8Pack p;
  p.h[0] = (__bf16)a.x; p.h[1] = (__bf16)a.y; p.h[2] = (__bf16)a.z; p.h[3] = (__bf16)a.w;
  p.h[4] = (__bf16)b.x; p.h[5] = (__bf16)b.y; p.h[6] = (__bf16)b.z; p.h[7] = (__bf16)b.w;
  reinterpret_cast<uint4*>(xb)[i] = p.u;
}

// ---------------------------------------------------------------------------
// 2) pack conv_w (D,D,3) f32 into WMMA bf16 B-fragment order.
// K index = k*2048 + din (three slabs). Fragment layout (contiguous k order, as A; 16-bit B 32x16):
//   lanes 0-15 hold K = kq*32 + 0..15 (elem e -> K=e), lanes 16-31 hold K+16.
//   lane's n = nt*16 + (lane&15).
// Flat: frag lane block = ((nt*192 + kq)*32 + lane)*16 elements (32B contiguous).
// ---------------------------------------------------------------------------
__global__ void k_pack_convw(const float* __restrict__ w, unsigned short* __restrict__ wp) {
  int idx  = blockIdx.x * 256 + threadIdx.x;    // < 128*192*32 = 786432
  int lane = idx & 31;
  int kq   = (idx >> 5) % 192;
  int nt   = idx / (32 * 192);
  int n     = (nt << 4) + (lane & 15);
  int kbase = kq * 32 + ((lane >> 4) << 4);
  BF8Pack lo, hi;
#pragma unroll
  for (int e = 0; e < 16; ++e) {
    int K   = kbase + e;
    int k   = K >> 11;        // K / 2048
    int din = K & 2047;
    float v = w[(size_t)n * (Dd * 3) + (size_t)din * 3 + k];
    if (e < 8) lo.h[e] = (__bf16)v; else hi.h[e - 8] = (__bf16)v;
  }
  uint4* o = reinterpret_cast<uint4*>(wp) + (size_t)idx * 2;
  o[0] = lo.u; o[1] = hi.u;
}

// ---------------------------------------------------------------------------
// 3) pack cls_w (21,2048) into same fragment order, N padded to 32 (nt=0..1)
// ---------------------------------------------------------------------------
__global__ void k_pack_clsw(const float* __restrict__ w, unsigned short* __restrict__ cwp) {
  int idx  = blockIdx.x * 256 + threadIdx.x;   // < 2*64*32 = 4096
  int lane = idx & 31;
  int kq   = (idx >> 5) & 63;
  int nt   = idx >> 11;
  int n     = (nt << 4) + (lane & 15);
  int kbase = kq * 32 + ((lane >> 4) << 4);
  BF8Pack lo, hi;
#pragma unroll
  for (int e = 0; e < 16; ++e) {
    int K = kbase + e;
    float v = (n < NC) ? w[(size_t)n * Dd + K] : 0.0f;
    if (e < 8) lo.h[e] = (__bf16)v; else hi.h[e - 8] = (__bf16)v;
  }
  uint4* o = reinterpret_cast<uint4*>(cwp) + (size_t)idx * 2;
  o[0] = lo.u; o[1] = hi.u;
}

// ---------------------------------------------------------------------------
// 4) conv as GEMM with WMMA bf16.
//    Block (8 waves) shares one 16-row M-tile; the 18 input rows (t0-1..t0+16,
//    covering all three taps, boundary rows zero-filled) are staged into LDS
//    via gfx1250 async DMA (global_load_async_to_lds_b128 + s_wait_asynccnt).
//    Rows padded by 8 bf16 (row stride 4112B = 1028 dwords) so the 16 A-rows
//    of a fragment land in distinct LDS bank groups.
//    Each wave: 16x64 output tile, 192 K-chunks x 4 WMMA; A from LDS (2x
//    ds_load_b128/lane/chunk), B streamed from packed weights (L2-resident).
//    Fused bias + ReLU, store emb bf16.
// ---------------------------------------------------------------------------
#define APAD   8
#define AROW4  ((Dd + APAD) / 8)   // 257 uint4 per staged row

__global__ void __launch_bounds__(256, 1)
k_conv_wmma(const unsigned short* __restrict__ xbu, const unsigned short* __restrict__ wpu,
            const float* __restrict__ convb, unsigned short* __restrict__ embu) {
  __shared__ __align__(16) uint4 sA4[18 * AROW4];   // 73.9 KB

  const uint4* xb4 = reinterpret_cast<const uint4*>(xbu);
  const uint4* wp4 = reinterpret_cast<const uint4*>(wpu);
  __bf16*      emb = reinterpret_cast<__bf16*>(embu);

  int tid  = threadIdx.x;
  int wid  = tid >> 5;
  int lane = tid & 31;
  int mt = blockIdx.x >> 2;                   // 0..2047  M-tile (16 rows)
  int ng = ((blockIdx.x & 3) << 3) + wid;     // 0..31    N-group (64 cols)
  int b  = mt >> 6;
  int t0 = (mt & 63) << 4;

  // ---- stage A: rows t0-1 .. t0+16, one 16B chunk per thread per row ----
#pragma unroll 2
  for (int r = 0; r < 18; ++r) {
    int tr = t0 - 1 + r;
    int ldst = r * AROW4 + tid;               // uint4 index in LDS
    if ((unsigned)tr < (unsigned)Tt) {
      size_t gsrc = ((size_t)(b * Tt + tr) * Dd) / 8 + tid;
#if defined(HAVE_ASYNC_LDS)
      __builtin_amdgcn_global_load_async_to_lds_b128(
          (gv4i*)(__attribute__((address_space(1))) void*)(xb4 + gsrc),
          (lv4i*)(__attribute__((address_space(3))) void*)(sA4 + ldst),
          0, 0);
#else
      sA4[ldst] = xb4[gsrc];
#endif
    } else {
      sA4[ldst] = make_uint4(0u, 0u, 0u, 0u);
    }
  }
  wait_async_zero();
  __syncthreads();

  int laneHi = lane >> 4;
  int row    = lane & 15;
  int n0 = ng << 6;

  v8f acc[4] = {};

  for (int k = 0; k < 3; ++k) {
    const uint4* arow = sA4 + (size_t)(row + k) * AROW4;   // LDS row for this lane
    for (int dq = 0; dq < 64; ++dq) {
      int kq = k * 64 + dq;
      Frag32 fa;
      const uint4* p = arow + (dq << 2);      // 32 bf16 = 4 uint4 per K-chunk
      fa.u[0] = p[2 * laneHi];                // K 0..7  (lo lanes) / 16..23 (hi lanes)
      fa.u[1] = p[2 * laneHi + 1];            // K 8..15 / 24..31
#pragma unroll
      for (int ns = 0; ns < 4; ++ns) {
        Frag32 fb;
        size_t fi = ((size_t)((ng * 4 + ns) * 192 + kq)) * 64 + (size_t)lane * 2;
        fb.u[0] = wp4[fi];
        fb.u[1] = wp4[fi + 1];
        acc[ns] = __builtin_amdgcn_wmma_f32_16x16x32_bf16(
            false, fa.v, false, fb.v, (short)0, acc[ns], false, false);
      }
    }
  }

#pragma unroll
  for (int ns = 0; ns < 4; ++ns) {
    int n = n0 + (ns << 4) + row;             // C/D layout: lane%16 = N column
    float bias = convb[n];
#pragma unroll
    for (int e = 0; e < 8; ++e) {
      int r = t0 + e + (laneHi << 3);         // vgpr e -> M row (+8 for hi lanes)
      float v = acc[ns][e] + bias;
      v = v > 0.0f ? v : 0.0f;
      emb[(size_t)(b * Tt + r) * Dd + n] = (__bf16)v;
    }
  }
}

// ---------------------------------------------------------------------------
// 5) attention 1x1 conv (D->2) + 2-way softmax -> temp_att (ws + out)
// ---------------------------------------------------------------------------
__global__ void k_att(const unsigned short* __restrict__ embu, const float* __restrict__ attw,
                      const float* __restrict__ attb, float* __restrict__ attws,
                      float* __restrict__ out) {
  int idx = blockIdx.x * 256 + threadIdx.x;   // < 32768 = B*T
  const uint4* rowp = reinterpret_cast<const uint4*>(embu) + (size_t)idx * (Dd / 8);
  float a0 = 0.0f, a1 = 0.0f;
  for (int q = 0; q < Dd / 8; ++q) {
    BF8Pack p; p.u = rowp[q];
#pragma unroll
    for (int e = 0; e < 8; ++e) {
      float ev = (float)p.h[e];
      int d = q * 8 + e;
      a0 += ev * attw[d];
      a1 += ev * attw[Dd + d];
    }
  }
  a0 += attb[0]; a1 += attb[1];
  float m  = fmaxf(a0, a1);
  float e0 = __expf(a0 - m), e1 = __expf(a1 - m);
  float inv = 1.0f / (e0 + e1);
  float fg = e0 * inv, bg = e1 * inv;
  attws[idx * 2 + 0] = fg;
  attws[idx * 2 + 1] = bg;
  out[OFF_ATT + idx * 2 + 0] = fg;
  out[OFF_ATT + idx * 2 + 1] = bg;
}

// ---------------------------------------------------------------------------
// 6) cas GEMM with WMMA bf16: M=B*T, N=21 (padded 32), K=2048. One wave per M-tile.
// ---------------------------------------------------------------------------
__global__ void __launch_bounds__(256, 1)
k_cas_wmma(const unsigned short* __restrict__ embu, const unsigned short* __restrict__ cwpu,
           const float* __restrict__ clsb, float* __restrict__ cas) {
  const __bf16* emb = reinterpret_cast<const __bf16*>(embu);
  const uint4*  cw4 = reinterpret_cast<const uint4*>(cwpu);
  int wid  = threadIdx.x >> 5;
  int lane = threadIdx.x & 31;
  int mt = blockIdx.x * 8 + wid;              // 0..2047
  int b  = mt >> 6;
  int t0 = (mt & 63) << 4;
  int laneHi = lane >> 4;
  int row    = lane & 15;
  int t = t0 + row;
  const uint4* arow = reinterpret_cast<const uint4*>(emb + (size_t)(b * Tt + t) * Dd);

  v8f acc[2] = {};
  for (int kq = 0; kq < 64; ++kq) {
    Frag32 fa;
    const uint4* p = arow + (kq << 2);
    fa.u[0] = p[2 * laneHi];
    fa.u[1] = p[2 * laneHi + 1];
#pragma unroll
    for (int nt = 0; nt < 2; ++nt) {
      Frag32 fb;
      size_t fi = ((size_t)(nt * 64 + kq)) * 64 + (size_t)lane * 2;
      fb.u[0] = cw4[fi];
      fb.u[1] = cw4[fi + 1];
      acc[nt] = __builtin_amdgcn_wmma_f32_16x16x32_bf16(
          false, fa.v, false, fb.v, (short)0, acc[nt], false, false);
    }
  }
#pragma unroll
  for (int nt = 0; nt < 2; ++nt) {
    int n = (nt << 4) + row;
    if (n < NC) {
      float bias = clsb[n];
#pragma unroll
      for (int e = 0; e < 8; ++e) {
        int r = t0 + e + (laneHi << 3);
        cas[(size_t)(b * Tt + r) * NC + n] = acc[nt][e] + bias;
      }
    }
  }
}

// ---------------------------------------------------------------------------
// 7) per-(b,t) softmax over classes for cas, fg_cas, bg_cas
// ---------------------------------------------------------------------------
__global__ void k_softmax_cas(const float* __restrict__ cas, const float* __restrict__ attws,
                              float* __restrict__ out) {
  int idx = blockIdx.x * 256 + threadIdx.x;   // < 32768
  const float* cr = cas + (size_t)idx * NC;
  float fga = attws[idx * 2 + 0];
  float bga = attws[idx * 2 + 1];
  float v[NC];
#pragma unroll
  for (int c = 0; c < NC; ++c) v[c] = cr[c];
  float m0 = -1e30f, m1 = -1e30f, m2 = -1e30f;
#pragma unroll
  for (int c = 0; c < NC; ++c) {
    m0 = fmaxf(m0, v[c]);
    m1 = fmaxf(m1, v[c] * fga);
    m2 = fmaxf(m2, v[c] * bga);
  }
  float s0 = 0.0f, s1 = 0.0f, s2 = 0.0f;
#pragma unroll
  for (int c = 0; c < NC; ++c) {
    s0 += __expf(v[c] - m0);
    s1 += __expf(v[c] * fga - m1);
    s2 += __expf(v[c] * bga - m2);
  }
  float i0 = 1.0f / s0, i1 = 1.0f / s1, i2 = 1.0f / s2;
  float* o0 = out + OFF_CAS + (size_t)idx * NC;
  float* o1 = out + OFF_FGC + (size_t)idx * NC;
  float* o2 = out + OFF_BGC + (size_t)idx * NC;
#pragma unroll
  for (int c = 0; c < NC; ++c) {
    o0[c] = __expf(v[c] - m0) * i0;
    o1[c] = __expf(v[c] * fga - m1) * i1;
    o2[c] = __expf(v[c] * bga - m2) * i2;
  }
}

// ---------------------------------------------------------------------------
// 8) top-k mean per (b,c): bitonic sort of 1024 values in LDS, mean of top k.
// ---------------------------------------------------------------------------
__device__ void bitonic_desc(float* s, int tid) {
  for (int k = 2; k <= Tt; k <<= 1) {
    for (int j = k >> 1; j > 0; j >>= 1) {
      int ixj = tid ^ j;
      if (ixj > tid) {
        float a = s[tid], b = s[ixj];
        bool dir  = ((tid & k) == 0);              // overall descending
        bool swap = dir ? (a < b) : (a > b);
        if (swap) { s[tid] = b; s[ixj] = a; }
      }
      __syncthreads();
    }
  }
}

__device__ float topk_mean(float* s, int tid, int kk) {
  s[tid] = (tid < kk) ? s[tid] : 0.0f;
  __syncthreads();
  for (int st = Tt / 2; st > 0; st >>= 1) {
    if (tid < st) s[tid] += s[tid + st];
    __syncthreads();
  }
  float r = s[0] / (float)kk;
  __syncthreads();
  return r;
}

__global__ void __launch_bounds__(1024)
k_topk(const float* __restrict__ cas, const float* __restrict__ attws,
       float* __restrict__ fgm, float* __restrict__ bgm) {
  __shared__ float sfg[Tt];
  __shared__ float sbg[Tt];
  int c = blockIdx.x % NC;
  int b = blockIdx.x / NC;
  int tid = threadIdx.x;
  float v = cas[(size_t)(b * Tt + tid) * NC + c];
  sfg[tid] = v * attws[(b * Tt + tid) * 2 + 0];
  sbg[tid] = v * attws[(b * Tt + tid) * 2 + 1];
  __syncthreads();
  bitonic_desc(sfg, tid);
  float fmean = topk_mean(sfg, tid, Tt / 8);       // fg_k = 128
  bitonic_desc(sbg, tid);
  float bmean = topk_mean(sbg, tid, Tt / 3);       // bg_k = 341
  if (tid == 0) {
    fgm[b * NC + c] = fmean;
    bgm[b * NC + c] = bmean;
  }
}

// ---------------------------------------------------------------------------
// 9) final per-batch class softmax (wave32 shuffle reduce)
// ---------------------------------------------------------------------------
__global__ void k_final(const float* __restrict__ fgm, const float* __restrict__ bgm,
                        float* __restrict__ out) {
  int b = blockIdx.x;
  int c = threadIdx.x;  // 32 threads
  for (int which = 0; which < 2; ++which) {
    const float* src = which ? bgm : fgm;
    float v = (c < NC) ? src[b * NC + c] : -1e30f;
    float m = v;
    for (int o = 16; o > 0; o >>= 1) m = fmaxf(m, __shfl_xor(m, o, 32));
    float e = (c < NC) ? __expf(v - m) : 0.0f;
    float s = e;
    for (int o = 16; o > 0; o >>= 1) s += __shfl_xor(s, o, 32);
    if (c < NC) out[(which ? OFF_BG : OFF_FG) + b * NC + c] = e / s;
  }
}

// ---------------------------------------------------------------------------
extern "C" void kernel_launch(void* const* d_in, const int* in_sizes, int n_in,
                              void* d_out, int out_size, void* d_ws, size_t ws_size,
                              hipStream_t stream) {
  const float* x     = (const float*)d_in[0];
  const float* convw = (const float*)d_in[1];
  const float* convb = (const float*)d_in[2];
  const float* attw  = (const float*)d_in[3];
  const float* attb  = (const float*)d_in[4];
  const float* clsw  = (const float*)d_in[5];
  const float* clsb  = (const float*)d_in[6];
  float* out = (float*)d_out;
  char* ws = (char*)d_ws;

  unsigned short* xb  = (unsigned short*)(ws + (size_t)0);           // 134,217,728 B
  unsigned short* emb = (unsigned short*)(ws + (size_t)134217728);   // 134,217,728 B
  unsigned short* wp  = (unsigned short*)(ws + (size_t)268435456);   //  25,165,824 B
  unsigned short* cwp = (unsigned short*)(ws + (size_t)293601280);   //     131,072 B
  float* cas   = (float*)(ws + (size_t)293732352);                   //   2,752,512 B
  float* attws = (float*)(ws + (size_t)296484864);                   //     262,144 B
  float* fgm   = (float*)(ws + (size_t)296747008);                   //       2,688 B
  float* bgm   = (float*)(ws + (size_t)296749696);                   //       2,688 B

  k_cast_x<<<32768, 256, 0, stream>>>(x, xb);
  k_pack_convw<<<3072, 256, 0, stream>>>(convw, wp);
  k_pack_clsw<<<16, 256, 0, stream>>>(clsw, cwp);
  k_conv_wmma<<<8192, 256, 0, stream>>>(xb, wp, convb, emb);
  k_att<<<128, 256, 0, stream>>>(emb, attw, attb, attws, out);
  k_cas_wmma<<<256, 256, 0, stream>>>(emb, cwp, clsb, cas);
  k_softmax_cas<<<128, 256, 0, stream>>>(cas, attws, out);
  k_topk<<<672, 1024, 0, stream>>>(cas, attws, fgm, bgm);
  k_final<<<32, 32, 0, stream>>>(fgm, bgm, out);
}
